// DynamicsPredictor_39900246180609
// MI455X (gfx1250) — matrix operand order repaired and checked by reading, not compiled, not run on hardware
//
#include <hip/hip_runtime.h>
#include <hip/hip_bf16.h>

#define BN     4
#define NN     2048
#define NPP    2000
#define NRELN  8192
#define NFN    128

typedef _Float16 h16;
typedef _Float16 v16h __attribute__((ext_vector_type(16)));
typedef _Float16 v8h  __attribute__((ext_vector_type(8)));
typedef float    v8f  __attribute__((ext_vector_type(8)));

// ---------- convert f32 weight (Ksrc x 128) -> f16, zero-padded to Kpad, in
// ---------- WMMA B-fragment-friendly layout: elem(k,col) at ((kt*2+g)*128+col)*16+i
// ---------- where kt=k/32, g=(k/16)&1, i=k&15
__global__ void conv_w_kernel(const float* __restrict__ w, int Ksrc, int Kpad,
                              h16* __restrict__ dst) {
  int e = blockIdx.x * blockDim.x + threadIdx.x;
  if (e >= Kpad * 128) return;
  int k = e >> 7, col = e & 127;
  float v = (k < Ksrc) ? w[k * 128 + col] : 0.0f;
  int kt = k >> 5, g = (k >> 4) & 1, i = k & 15;
  dst[((size_t)((kt * 2 + g) * 128 + col)) * 16 + i] = (h16)v;
}

// ---------- one streaming pass over dense one-hot Rr/Rs -> index arrays ----------
__global__ void extract_idx_kernel(const float* __restrict__ Rr, const float* __restrict__ Rs,
                                   int* __restrict__ ri, int* __restrict__ si) {
  unsigned e = blockIdx.x * blockDim.x + threadIdx.x;   // exactly B*NREL*N threads
  float a = Rr[e];
  float b = Rs[e];
  unsigned row = e >> 11, col = e & 2047;               // N = 2048
  if (a > 0.5f) ri[row] = (int)col;
  if (b > 0.5f) si[row] = (int)col;
}

// ---------- build p_in (B*N x 32 f16, cols 25..31 zero-padded) ----------
__global__ void build_pin_kernel(const float* __restrict__ state, const float* __restrict__ attrs,
                                 const float* __restrict__ action, const float* __restrict__ den,
                                 const float* __restrict__ matenc, const float* __restrict__ phys,
                                 h16* __restrict__ pin) {
  int e = blockIdx.x * blockDim.x + threadIdx.x;
  if (e >= BN * NN) return;
  int b = e >> 11, n = e & 2047;
  h16* o = pin + (size_t)e * 32;
  o[0] = (h16)attrs[e * 2 + 0];
  o[1] = (h16)attrs[e * 2 + 1];
  #pragma unroll
  for (int h = 0; h < 4; ++h)
    #pragma unroll
    for (int d = 0; d < 3; ++d) {
      float sh;
      if (h < 3) sh = state[((b * 4 + h + 1) * NN + n) * 3 + d] - state[((b * 4 + h) * NN + n) * 3 + d];
      else       sh = state[((b * 4 + 3) * NN + n) * 3 + d];
      o[2 + h * 3 + d] = (h16)sh;
    }
  bool real = (n < NPP);
  #pragma unroll
  for (int m = 0; m < 6; ++m) o[14 + m] = real ? (h16)matenc[b * 6 + m] : (h16)0.0f;
  o[20] = real ? (h16)phys[b] : (h16)0.0f;
  o[21] = (h16)action[e * 3 + 0];
  o[22] = (h16)action[e * 3 + 1];
  o[23] = (h16)action[e * 3 + 2];
  o[24] = real ? (h16)den[b] : (h16)0.0f;
  #pragma unroll
  for (int c = 25; c < 32; ++c) o[c] = (h16)0.0f;
}

// ---------- build rel_in (B*NREL x 96 f16, cols 68..95 zero) ----------
__global__ void build_relin_kernel(const h16* __restrict__ pin, const int* __restrict__ ridx,
                                   const int* __restrict__ sidx, const float* __restrict__ pinst,
                                   h16* __restrict__ relin) {
  int e = blockIdx.x * blockDim.x + threadIdx.x;
  if (e >= BN * NRELN) return;
  int b = e >> 13;                                      // NREL = 8192
  int ri = ridx[e], si = sidx[e];
  const h16* pr = pin + (size_t)(b * NN + ri) * 32;
  const h16* ps = pin + (size_t)(b * NN + si) * 32;
  h16* o = relin + (size_t)e * 96;
  #pragma unroll
  for (int i = 0; i < 25; ++i) { o[i] = pr[i]; o[25 + i] = ps[i]; }
  o[50] = pr[0]; o[51] = pr[1]; o[52] = ps[0]; o[53] = ps[1];
  float gd = 0.0f;
  #pragma unroll
  for (int k = 0; k < 8; ++k) {
    float gr = (ri < NPP) ? pinst[(b * NPP + ri) * 8 + k] : 0.0f;
    float gs = (si < NPP) ? pinst[(b * NPP + si) * 8 + k] : 0.0f;
    gd += fabsf(gr - gs);
  }
  o[54] = (h16)gd;
  #pragma unroll
  for (int j = 0; j < 12; ++j) o[55 + j] = (h16)((float)pr[2 + j] - (float)ps[2 + j]);
  o[67] = (h16)((float)pr[24] - (float)ps[24]);
  #pragma unroll
  for (int c = 68; c < 96; ++c) o[c] = (h16)0.0f;
}

// ---------- WMMA GEMM: A(M x K, lda) * W(K x 128) + bias, ReLU epilogue.
// Modes: RESID adds f16 residual (ld 128); SCATTER does atomic scatter-add into agg.
// One wave = one 16x16 tile; 8 waves/block cover the full 128-wide N strip.
template<bool RESID, bool SCATTER>
__global__ __launch_bounds__(256) void gemm_wmma(
    const h16* __restrict__ A, int lda,
    const h16* __restrict__ Wsw, const float* __restrict__ bias, int K,
    h16* __restrict__ Out, int ldo,
    const h16* __restrict__ Resid,
    const int* __restrict__ scat_idx, float* __restrict__ agg) {
  const int wave  = threadIdx.x >> 5;
  const int lane  = threadIdx.x & 31;
  const int mBase = blockIdx.x << 4;
  const int nBase = wave << 4;
  const int col16 = lane & 15;
  const int half  = lane >> 4;

  v8f acc = {};
  const h16* aRow = A + (size_t)(mBase + col16) * lda + half * 16;     // A frag: row=lane%16, K-half by lane group
  const h16* wCol = Wsw + ((size_t)(half * 128 + nBase + col16)) * 16; // B frag: pre-swizzled contiguous
  const int nk = K >> 5;
  for (int kt = 0; kt < nk; ++kt) {
    v8h alo = *(const v8h*)(aRow + kt * 32);
    v8h ahi = *(const v8h*)(aRow + kt * 32 + 8);
    const v8h* wp = (const v8h*)(wCol + (size_t)kt * (2 * 128 * 16));
    v8h blo = wp[0];
    v8h bhi = wp[1];
    v16h a = __builtin_shufflevector(alo, ahi, 0,1,2,3,4,5,6,7,8,9,10,11,12,13,14,15);
    v16h b = __builtin_shufflevector(blo, bhi, 0,1,2,3,4,5,6,7,8,9,10,11,12,13,14,15);
    acc = __builtin_amdgcn_wmma_f32_16x16x32_f16(false, a, false, b, (short)0, acc, false, false);
  }
  const int gcol = nBase + col16;
  const float bv = bias[gcol];
  #pragma unroll
  for (int v = 0; v < 8; ++v) {
    int grow = mBase + half * 8 + v;                    // C layout: M = vgpr + 8*(lane>=16)
    float val = acc[v] + bv;
    if (RESID) val += (float)Resid[(size_t)grow * 128 + gcol];
    val = fmaxf(val, 0.0f);
    if (SCATTER) {
      int bb  = grow >> 13;                             // batch = row / NREL
      int tgt = (bb << 11) + scat_idx[grow];            // (b*N + recv_idx)
      atomicAdd(&agg[(size_t)tgt * 128 + gcol], val);
    } else {
      Out[(size_t)grow * ldo + gcol] = (h16)val;
    }
  }
}

// ---------- small elementwise helpers ----------
__global__ void copy_strided_h(const h16* __restrict__ src, int sld,
                               h16* __restrict__ dst, int dld, int rows) {
  int e = blockIdx.x * blockDim.x + threadIdx.x;
  if (e >= rows * 128) return;
  int r = e >> 7, c = e & 127;
  dst[(size_t)r * dld + c] = src[(size_t)r * sld + c];
}

__global__ void zero_f_kernel(float* __restrict__ p, int n) {
  int e = blockIdx.x * blockDim.x + threadIdx.x;
  if (e < n) p[e] = 0.0f;
}

__global__ void agg2pcat_kernel(const float* __restrict__ agg, h16* __restrict__ pcat) {
  int e = blockIdx.x * blockDim.x + threadIdx.x;
  if (e >= BN * NN * 128) return;
  int r = e >> 7, c = e & 127;
  pcat[(size_t)r * 256 + 128 + c] = (h16)agg[e];
}

__global__ void fill_eres_kernel(const h16* __restrict__ eff, const int* __restrict__ ridx,
                                 const int* __restrict__ sidx, h16* __restrict__ relcat) {
  int e = blockIdx.x * blockDim.x + threadIdx.x;
  if (e >= BN * NRELN * 256) return;
  int row = e >> 8, c = e & 255;
  int b = row >> 13;
  int idx = (c < 128) ? ridx[row] : sidx[row];
  relcat[(size_t)row * 384 + 128 + c] = eff[(size_t)(b * NN + idx) * 128 + (c & 127)];
}

// ---------- final 128->3 head + clip + position update ----------
__global__ void final_pred_kernel(const h16* __restrict__ x, const float* __restrict__ w2,
                                  const float* __restrict__ b2, const float* __restrict__ state,
                                  float* __restrict__ out) {
  int e = blockIdx.x * blockDim.x + threadIdx.x;
  if (e >= BN * NPP * 3) return;
  int d = e % 3;
  int p = (e / 3) % NPP;
  int b = e / (3 * NPP);
  const h16* xr = x + (size_t)(b * NPP + p) * 128;
  float s = b2[d];
  #pragma unroll 4
  for (int k = 0; k < 128; ++k) s += (float)xr[k] * w2[k * 3 + d];
  out[BN * NPP * 3 + e] = s;                                     // motion (unclipped)
  float c = fminf(fmaxf(s, -100.0f), 100.0f);
  out[e] = state[((b * 4 + 3) * NN + p) * 3 + d] + c;            // pred_pos
}

extern "C" void kernel_launch(void* const* d_in, const int* in_sizes, int n_in,
                              void* d_out, int out_size, void* d_ws, size_t ws_size,
                              hipStream_t stream) {
  const float* state  = (const float*)d_in[0];
  const float* attrs  = (const float*)d_in[1];
  const float* Rr     = (const float*)d_in[2];
  const float* Rs     = (const float*)d_in[3];
  const float* pinst  = (const float*)d_in[4];
  const float* action = (const float*)d_in[5];
  const float* den    = (const float*)d_in[6];
  const float* matenc = (const float*)d_in[7];
  const float* phys   = (const float*)d_in[8];
  // params in setup_inputs() insertion order: pe, re, rp, pp, pred
  const float* pew0 = (const float*)d_in[9],  *peb0 = (const float*)d_in[10];
  const float* pew1 = (const float*)d_in[11], *peb1 = (const float*)d_in[12];
  const float* pew2 = (const float*)d_in[13], *peb2 = (const float*)d_in[14];
  const float* rew0 = (const float*)d_in[15], *reb0 = (const float*)d_in[16];
  const float* rew1 = (const float*)d_in[17], *reb1 = (const float*)d_in[18];
  const float* rew2 = (const float*)d_in[19], *reb2 = (const float*)d_in[20];
  const float* rpw  = (const float*)d_in[21], *rpb  = (const float*)d_in[22];
  const float* ppw  = (const float*)d_in[23], *ppb  = (const float*)d_in[24];
  const float* pdw0 = (const float*)d_in[25], *pdb0 = (const float*)d_in[26];
  const float* pdw1 = (const float*)d_in[27], *pdb1 = (const float*)d_in[28];
  const float* pdw2 = (const float*)d_in[29], *pdb2 = (const float*)d_in[30];
  float* out = (float*)d_out;

  char* ws = (char*)d_ws;
  size_t off = 0;
  auto alloc = [&](size_t bytes) { size_t o = off; off = (off + bytes + 255) & ~(size_t)255; return o; };
  int*  ridx   = (int*)(ws + alloc((size_t)BN * NRELN * 4));
  int*  sidx   = (int*)(ws + alloc((size_t)BN * NRELN * 4));
  h16*  pin    = (h16*)(ws + alloc((size_t)BN * NN * 32 * 2));
  h16*  relin  = (h16*)(ws + alloc((size_t)BN * NRELN * 96 * 2));
  h16*  relcat = (h16*)(ws + alloc((size_t)BN * NRELN * 384 * 2));
  h16*  pcat   = (h16*)(ws + alloc((size_t)BN * NN * 256 * 2));
  h16*  effb   = (h16*)(ws + alloc((size_t)BN * NN * 128 * 2));
  h16*  t0     = (h16*)(ws + alloc((size_t)BN * NRELN * 128 * 2));
  h16*  t1     = (h16*)(ws + alloc((size_t)BN * NRELN * 128 * 2));
  float* agg   = (float*)(ws + alloc((size_t)BN * NN * 128 * 4));
  h16* wpe0 = (h16*)(ws + alloc((size_t)32  * 128 * 2));
  h16* wpe1 = (h16*)(ws + alloc((size_t)128 * 128 * 2));
  h16* wpe2 = (h16*)(ws + alloc((size_t)128 * 128 * 2));
  h16* wre0 = (h16*)(ws + alloc((size_t)96  * 128 * 2));
  h16* wre1 = (h16*)(ws + alloc((size_t)128 * 128 * 2));
  h16* wre2 = (h16*)(ws + alloc((size_t)128 * 128 * 2));
  h16* wrp  = (h16*)(ws + alloc((size_t)384 * 128 * 2));
  h16* wpp  = (h16*)(ws + alloc((size_t)256 * 128 * 2));
  h16* wpd0 = (h16*)(ws + alloc((size_t)128 * 128 * 2));
  h16* wpd1 = (h16*)(ws + alloc((size_t)128 * 128 * 2));
  (void)in_sizes; (void)n_in; (void)out_size; (void)ws_size;

  auto cw = [&](const float* src, int Ksrc, int Kpad, h16* dst) {
    conv_w_kernel<<<(Kpad * 128 + 255) / 256, 256, 0, stream>>>(src, Ksrc, Kpad, dst);
  };
  cw(pew0, 25, 32, wpe0);  cw(pew1, 128, 128, wpe1);  cw(pew2, 128, 128, wpe2);
  cw(rew0, 68, 96, wre0);  cw(rew1, 128, 128, wre1);  cw(rew2, 128, 128, wre2);
  cw(rpw, 384, 384, wrp);  cw(ppw, 256, 256, wpp);
  cw(pdw0, 128, 128, wpd0); cw(pdw1, 128, 128, wpd1);

  // one-hot -> index (single streaming pass over the 2x256MB matrices)
  extract_idx_kernel<<<(BN * NRELN * NN) / 256, 256, 0, stream>>>(Rr, Rs, ridx, sidx);

  build_pin_kernel<<<(BN * NN + 255) / 256, 256, 0, stream>>>(state, attrs, action, den, matenc, phys, pin);
  build_relin_kernel<<<(BN * NRELN + 255) / 256, 256, 0, stream>>>(pin, ridx, sidx, pinst, relin);

  const int gP = (BN * NN) / 16;     // 512 tiles-of-16 rows (particle GEMMs)
  const int gR = (BN * NRELN) / 16;  // 2048 (relation GEMMs)

  // particle encoder -> pcat cols 0..127 ; eff = penc
  gemm_wmma<false,false><<<gP, 256, 0, stream>>>(pin, 32,  wpe0, peb0, 32,  t0,   128, nullptr, nullptr, nullptr);
  gemm_wmma<false,false><<<gP, 256, 0, stream>>>(t0,  128, wpe1, peb1, 128, t1,   128, nullptr, nullptr, nullptr);
  gemm_wmma<false,false><<<gP, 256, 0, stream>>>(t1,  128, wpe2, peb2, 128, pcat, 256, nullptr, nullptr, nullptr);
  copy_strided_h<<<(BN * NN * 128 + 255) / 256, 256, 0, stream>>>(pcat, 256, effb, 128, BN * NN);

  // relation encoder -> relcat cols 0..127
  gemm_wmma<false,false><<<gR, 256, 0, stream>>>(relin, 96, wre0, reb0, 96,  t0,     128, nullptr, nullptr, nullptr);
  gemm_wmma<false,false><<<gR, 256, 0, stream>>>(t0,   128, wre1, reb1, 128, t1,     128, nullptr, nullptr, nullptr);
  gemm_wmma<false,false><<<gR, 256, 0, stream>>>(t1,   128, wre2, reb2, 128, relcat, 384, nullptr, nullptr, nullptr);

  // 3 propagation steps
  for (int step = 0; step < 3; ++step) {
    fill_eres_kernel<<<(BN * NRELN * 256 + 255) / 256, 256, 0, stream>>>(effb, ridx, sidx, relcat);
    zero_f_kernel<<<(BN * NN * 128 + 255) / 256, 256, 0, stream>>>(agg, BN * NN * 128);
    // rel MLP + ReLU fused with atomic scatter-add into agg
    gemm_wmma<false,true><<<gR, 256, 0, stream>>>(relcat, 384, wrp, rpb, 384, nullptr, 0, nullptr, ridx, agg);
    agg2pcat_kernel<<<(BN * NN * 128 + 255) / 256, 256, 0, stream>>>(agg, pcat);
    // particle MLP + residual + ReLU, in-place eff update
    gemm_wmma<true,false><<<gP, 256, 0, stream>>>(pcat, 256, wpp, ppb, 256, effb, 128, effb, nullptr, nullptr);
  }

  // prediction head (per batch so rows stay 16-aligned: 2000 = 125*16)
  for (int b = 0; b < BN; ++b) {
    gemm_wmma<false,false><<<NPP / 16, 256, 0, stream>>>(effb + (size_t)b * NN * 128, 128, wpd0, pdb0, 128,
                                                         t0 + (size_t)b * NPP * 128, 128, nullptr, nullptr, nullptr);
    gemm_wmma<false,false><<<NPP / 16, 256, 0, stream>>>(t0 + (size_t)b * NPP * 128, 128, wpd1, pdb1, 128,
                                                         t1 + (size_t)b * NPP * 128, 128, nullptr, nullptr, nullptr);
  }
  final_pred_kernel<<<(BN * NPP * 3 + 255) / 256, 256, 0, stream>>>(t1, pdw2, pdb2, state, out);
}
